// MixedScaleSparseTransformerBlock_41128606826827
// MI455X (gfx1250) — matrix operand order repaired and checked by reading, not compiled, not run on hardware
//
#include <hip/hip_runtime.h>
#include <hip/hip_bf16.h>

// ---------------------------------------------------------------------------
// MI455X (gfx1250) sparse windowed transformer block.
// - All dense math on V_WMMA_F32_16X16X32_F16 (wave32).
// - GEMM A-panels staged to LDS via GLOBAL_LOAD_ASYNC_TO_LDS_B128 (ASYNCcnt).
// - fp32 kept for LN, softmax statistics, scatter accumulation, residuals.
// ---------------------------------------------------------------------------

typedef _Float16 h16v __attribute__((ext_vector_type(16)));
typedef _Float16 h8v  __attribute__((ext_vector_type(8)));
typedef float    f8v  __attribute__((ext_vector_type(8)));

static constexpr int Npt = 120000;   // N voxels
static constexpr int Cc  = 128;      // channels
static constexpr int Wn  = 12000;    // windows
static constexpr int SQi = 16;       // queries / window
static constexpr int SKi = 64;       // keys / window
static constexpr int Hh  = 8;        // heads
static constexpr int Dd  = 16;       // head dim
static constexpr int FFd = 512;      // FFN hidden
static constexpr int PQ  = Wn * SQi; // 192000
static constexpr int PK  = Wn * SKi; // 768000

// ---- WMMA helpers ---------------------------------------------------------

__device__ __forceinline__ f8v wmma_f16(h16v a, h16v b, f8v c) {
    // D = A(16x32 f16) * B(32x16 f16) + C(16x16 f32)
    return __builtin_amdgcn_wmma_f32_16x16x32_f16(
        /*neg_a*/false, a, /*neg_b*/false, b, /*c_mod*/(short)0, c,
        /*reuse_a*/false, /*reuse_b*/false);
}

// A-fragment: lane(sub=lane>>4, m=lane&15) holds row m,
// halves [kk+sub*8 .. +7] and [kk+16+sub*8 .. +7].
// `row` must already point at  &A[m*lda + kk + sub*8].
__device__ __forceinline__ h16v load_a_frag(const _Float16* __restrict__ row) {
    h8v lo = *(const h8v*)(row);
    h8v hi = *(const h8v*)(row + 16);
    return __builtin_shufflevector(lo, hi,
        0,1,2,3,4,5,6,7,8,9,10,11,12,13,14,15);
}

// Async global->LDS 16B copy per lane (tracked by ASYNCcnt).
__device__ __forceinline__ void async_g2l_b128(unsigned lds_off, const void* gaddr) {
    asm volatile("global_load_async_to_lds_b128 %0, %1, off"
                 :: "v"(lds_off), "v"(gaddr) : "memory");
}
__device__ __forceinline__ void wait_asynccnt0() {
    asm volatile("s_wait_asynccnt 0" ::: "memory");
}

// ---- weight packing: fp32 W[K][Nt] -> f16 B-fragment tiles ----------------
// tile t = n_tile*(K/32)+k_tile; element (lane,i) = W[k_tile*32+(i<8 ? (lane>>4)*8+i : 16+(lane>>4)*8+i-8)][n_tile*16+(lane&15)]
__global__ __launch_bounds__(256) void pack_b_kernel(
    const float* __restrict__ Wt, _Float16* __restrict__ P, int K, int Nt)
{
    int idx = blockIdx.x * blockDim.x + threadIdx.x;
    if (idx >= K * Nt) return;
    int i      = idx & 15;
    int lane   = (idx >> 4) & 31;
    int tile   = idx >> 9;
    int ktiles = K >> 5;
    int k_tile = tile % ktiles;
    int n_tile = tile / ktiles;
    int kk = k_tile * 32 + (i < 8 ? (lane >> 4) * 8 + i : 16 + (lane >> 4) * 8 + (i - 8));
    int nn = n_tile * 16 + (lane & 15);
    P[idx] = (_Float16)Wt[(size_t)kk * Nt + nn];
}

// ---- LayerNorm 1: one wave per row ----------------------------------------
__global__ __launch_bounds__(256) void ln1_kernel(
    const float* __restrict__ x, const float* __restrict__ g,
    const float* __restrict__ b, float* __restrict__ out32, int nrows)
{
    int row  = blockIdx.x * 8 + (threadIdx.x >> 5);
    if (row >= nrows) return;
    int lane = threadIdx.x & 31;
    const float* xr = x + (size_t)row * Cc;
    float4 v = *(const float4*)(xr + lane * 4);
    float s = v.x + v.y + v.z + v.w;
    #pragma unroll
    for (int off = 16; off; off >>= 1) s += __shfl_xor(s, off, 32);
    float mean = s * (1.0f / Cc);
    float d0 = v.x - mean, d1 = v.y - mean, d2 = v.z - mean, d3 = v.w - mean;
    float vs = d0*d0 + d1*d1 + d2*d2 + d3*d3;
    #pragma unroll
    for (int off = 16; off; off >>= 1) vs += __shfl_xor(vs, off, 32);
    float rstd = rsqrtf(vs * (1.0f / Cc) + 1e-5f);
    float* orow = out32 + (size_t)row * Cc;
    int c0 = lane * 4;
    orow[c0+0] = d0 * rstd * g[c0+0] + b[c0+0];
    orow[c0+1] = d1 * rstd * g[c0+1] + b[c0+1];
    orow[c0+2] = d2 * rstd * g[c0+2] + b[c0+2];
    orow[c0+3] = d3 * rstd * g[c0+3] + b[c0+3];
}

// ---- pos-MLP (6->128 ReLU ->128 ReLU) + gather-add h[ind] -> f16 ----------
__global__ __launch_bounds__(256) void posgather_kernel(
    const float* __restrict__ pos,      // [P][6]
    const float* __restrict__ pw1,      // [6][128]
    const float* __restrict__ pb1,      // [128]
    const _Float16* __restrict__ pw2p,  // packed 128x128
    const float* __restrict__ pb2,      // [128]
    const float* __restrict__ h32,      // [N][128]
    const int* __restrict__ ind,        // [P]
    _Float16* __restrict__ out16)       // [P][128]
{
    __shared__ alignas(16) _Float16 t1[16 * Cc];
    const int pt0 = blockIdx.x * 16;
    const int tid = threadIdx.x;

    // layer 1: 16 threads per point, 8 columns each
    {
        int lp = tid >> 4;
        int c0 = (tid & 15) * 8;
        const float* p = pos + (size_t)(pt0 + lp) * 6;
        float pv0=p[0], pv1=p[1], pv2=p[2], pv3=p[3], pv4=p[4], pv5=p[5];
        #pragma unroll
        for (int c = 0; c < 8; ++c) {
            int col = c0 + c;
            float acc = pb1[col];
            acc += pv0 * pw1[0*Cc + col];
            acc += pv1 * pw1[1*Cc + col];
            acc += pv2 * pw1[2*Cc + col];
            acc += pv3 * pw1[3*Cc + col];
            acc += pv4 * pw1[4*Cc + col];
            acc += pv5 * pw1[5*Cc + col];
            t1[lp * Cc + col] = (_Float16)fmaxf(acc, 0.0f);
        }
    }
    __syncthreads();

    // layer 2 via WMMA: wave -> one 16x16 output tile
    const int lane = tid & 31, wave = tid >> 5;
    const int n0 = wave * 16;
    const int sub = lane >> 4, nn = lane & 15;
    f8v acc = {};
    #pragma unroll
    for (int kt = 0; kt < 4; ++kt) {
        const int kk = kt * 32;
        h16v a = load_a_frag(&t1[nn * Cc + kk + sub * 8]);
        h16v b = *(const h16v*)(pw2p + ((size_t)((n0 >> 4) * 4 + kt) * 32 + lane) * 16);
        acc = wmma_f16(a, b, acc);
    }
    const int col = n0 + nn;
    #pragma unroll
    for (int r = 0; r < 8; ++r) {
        int pt = pt0 + r + 8 * sub;
        float v = fmaxf(acc[r] + pb2[col], 0.0f)
                + h32[(size_t)ind[pt] * Cc + col];
        out16[(size_t)pt * Cc + col] = (_Float16)v;
    }
}

// ---- generic WMMA GEMM: out = A(f16 MxK) * packedB + bias -----------------
// A panel (16 rows x K, contiguous in memory) is staged to LDS with async
// global->LDS copies; all 8 waves feed WMMA from LDS.
// EPI 0: store f16   1: relu+store f16   2: masked atomic scatter   3: fp32 + resid
template<int EPI, int K>
__global__ __launch_bounds__(256) void gemm16(
    const _Float16* __restrict__ A, const _Float16* __restrict__ Bp,
    const float* __restrict__ bias, void* __restrict__ outp,
    int Nout,
    const int* __restrict__ ind, const unsigned char* __restrict__ mask,
    float* __restrict__ sums, const float* __restrict__ resid)
{
    __shared__ alignas(16) _Float16 sA[16 * K];
    const int tid  = threadIdx.x;
    const int row0 = blockIdx.x * 16;

    // stage the 16xK A panel (one contiguous 32*K-byte region) into LDS
    {
        constexpr int CHUNKS = (16 * K * 2) / 16;   // 16-byte chunks
        constexpr int PER_T  = CHUNKS / 256;
        const char* gsrc   = (const char*)(A + (size_t)row0 * K);
        unsigned    ldsbase = (unsigned)(size_t)(void*)sA;
        #pragma unroll
        for (int t = 0; t < PER_T; ++t) {
            int c = tid + t * 256;
            async_g2l_b128(ldsbase + (unsigned)c * 16, gsrc + (size_t)c * 16);
        }
        wait_asynccnt0();
    }
    __syncthreads();

    const int lane = tid & 31;
    const int wave = tid >> 5;
    const int n0   = blockIdx.y * 128 + wave * 16;   // always < Nout by launch
    const int sub  = lane >> 4, nn = lane & 15;
    constexpr int ktiles = K >> 5;
    const _Float16* btile = Bp + ((size_t)(n0 >> 4) * ktiles) * 512 + (size_t)lane * 16;
    const _Float16* arow  = &sA[nn * K + sub * 8];

    f8v acc = {};
    #pragma unroll
    for (int kt = 0; kt < ktiles; ++kt) {
        h16v a = load_a_frag(arow + kt * 32);
        h16v b = *(const h16v*)(btile + (size_t)kt * 512);
        acc = wmma_f16(a, b, acc);
    }

    const int col = n0 + nn;
    if (EPI == 0 || EPI == 1) {
        _Float16* O = (_Float16*)outp;
        #pragma unroll
        for (int r = 0; r < 8; ++r) {
            int row = row0 + r + 8 * sub;
            float v = acc[r] + bias[col];
            if (EPI == 1) v = fmaxf(v, 0.0f);
            O[(size_t)row * Nout + col] = (_Float16)v;
        }
    } else if (EPI == 2) {
        #pragma unroll
        for (int r = 0; r < 8; ++r) {
            int row = row0 + r + 8 * sub;
            if (!mask[row]) {
                float v = acc[r] + bias[col];
                atomicAdd(&sums[(size_t)ind[row] * Nout + col], v);
            }
        }
    } else { // EPI == 3
        float* O = (float*)outp;
        #pragma unroll
        for (int r = 0; r < 8; ++r) {
            int row = row0 + r + 8 * sub;
            O[(size_t)row * Nout + col] =
                resid[(size_t)row * Nout + col] + acc[r] + bias[col];
        }
    }
}

// ---- attention: one wave per (window, head) -------------------------------
__global__ __launch_bounds__(256) void attn_kernel(
    const _Float16* __restrict__ q16,   // [W*16][128]
    const _Float16* __restrict__ k16,   // [W*64][128]
    const _Float16* __restrict__ v16,   // [W*64][128]
    const unsigned char* __restrict__ kmask, // [W*64]
    _Float16* __restrict__ o16)         // [W*16][128]
{
    __shared__ alignas(16) float    sbuf[8][SQi * SKi];  // 32 KB
    __shared__ alignas(16) _Float16 abuf[8][SQi * SKi];  // 16 KB
    __shared__ float sinv[8][SQi];

    const int lane = threadIdx.x & 31, wave = threadIdx.x >> 5;
    const int task = blockIdx.x * 8 + wave;               // W*H / 8 blocks exactly
    const int w = task >> 3, h = task & 7;
    const int sub = lane >> 4, nn = lane & 15;

    const _Float16* qbase = q16 + ((size_t)w * SQi) * Cc + h * Dd;
    const _Float16* kbase = k16 + ((size_t)w * SKi) * Cc + h * Dd;
    const _Float16* vbase = v16 + ((size_t)w * SKi) * Cc + h * Dd;

    // Q A-fragment: K(reduction)=16 valid, upper 16 zero-padded
    h8v  z8  = {};
    h16v z16 = {};
    h16v aq  = sub ? z16 : *(const h16v*)(qbase + (size_t)nn * Cc);

    // S = Q * K^T  (4 column tiles of 16 keys)
    #pragma unroll
    for (int t = 0; t < 4; ++t) {
        const int ki = t * 16 + nn;
        h16v bk = *(const h16v*)(kbase + (size_t)ki * Cc); // k row: d=0..15
        if (sub) bk = z16;                                  // pad K=16..31
        f8v c = {};
        c = wmma_f16(aq, bk, c);
        const int km = kmask[(size_t)w * SKi + ki];
        #pragma unroll
        for (int r = 0; r < 8; ++r) {
            int qi = r + 8 * sub;
            sbuf[wave][qi * SKi + t * 16 + nn] = km ? -1e9f : c[r] * 0.25f;
        }
    }
    __syncthreads();

    // softmax: lanes 0..15 each own one query row
    if (lane < SQi) {
        const float* row = &sbuf[wave][lane * SKi];
        float mx = -3.4e38f;
        for (int j = 0; j < SKi; ++j) mx = fmaxf(mx, row[j]);
        float sum = 0.0f;
        for (int j = 0; j < SKi; ++j) {
            float e = __expf(row[j] - mx);
            sum += e;
            abuf[wave][lane * SKi + j] = (_Float16)e;  // e in [0,1]: safe in f16
        }
        sinv[wave][lane] = 1.0f / sum;
    }
    __syncthreads();

    // O = P * V   (K=64 -> two 16x16x32 WMMA steps)
    f8v oc = {};
    #pragma unroll
    for (int kt = 0; kt < 2; ++kt) {
        const int kk = kt * 32;
        h16v a = load_a_frag(&abuf[wave][nn * SKi + kk + sub * 8]);
        h16v bv;
        #pragma unroll
        for (int i = 0; i < 16; ++i)
            bv[i] = vbase[(size_t)(kk + (i < 8 ? sub * 8 + i : 16 + sub * 8 + (i - 8))) * Cc + nn];
        oc = wmma_f16(a, bv, oc);
    }
    #pragma unroll
    for (int r = 0; r < 8; ++r) {
        int qi = r + 8 * sub;
        float val = oc[r] * sinv[wave][qi];
        o16[((size_t)w * SQi + qi) * Cc + h * Dd + nn] = (_Float16)val;
    }
}

// ---- per-query validity count scatter -------------------------------------
__global__ __launch_bounds__(256) void cnt_kernel(
    const int* __restrict__ q_ind, const unsigned char* __restrict__ q_mask,
    float* __restrict__ cnt, int P)
{
    int i = blockIdx.x * blockDim.x + threadIdx.x;
    if (i >= P) return;
    if (!q_mask[i]) atomicAdd(&cnt[q_ind[i]], 1.0f);
}

// ---- residual average + LayerNorm 2 ---------------------------------------
__global__ __launch_bounds__(256) void resid_ln2_kernel(
    const float* __restrict__ feat, const float* __restrict__ sums,
    const float* __restrict__ cnt, const float* __restrict__ g,
    const float* __restrict__ b, float* __restrict__ y32,
    _Float16* __restrict__ h2_16, int nrows)
{
    int row  = blockIdx.x * 8 + (threadIdx.x >> 5);
    if (row >= nrows) return;
    int lane = threadIdx.x & 31;
    int c0   = lane * 4;
    float invc = 1.0f / fmaxf(cnt[row], 1.0f);
    float4 f = *(const float4*)(feat + (size_t)row * Cc + c0);
    float4 s = *(const float4*)(sums + (size_t)row * Cc + c0);
    float4 y;
    y.x = f.x + s.x * invc; y.y = f.y + s.y * invc;
    y.z = f.z + s.z * invc; y.w = f.w + s.w * invc;
    *(float4*)(y32 + (size_t)row * Cc + c0) = y;

    float sm = y.x + y.y + y.z + y.w;
    #pragma unroll
    for (int off = 16; off; off >>= 1) sm += __shfl_xor(sm, off, 32);
    float mean = sm * (1.0f / Cc);
    float d0 = y.x - mean, d1 = y.y - mean, d2 = y.z - mean, d3 = y.w - mean;
    float vs = d0*d0 + d1*d1 + d2*d2 + d3*d3;
    #pragma unroll
    for (int off = 16; off; off >>= 1) vs += __shfl_xor(vs, off, 32);
    float rstd = rsqrtf(vs * (1.0f / Cc) + 1e-5f);
    _Float16* orow = h2_16 + (size_t)row * Cc;
    orow[c0+0] = (_Float16)(d0 * rstd * g[c0+0] + b[c0+0]);
    orow[c0+1] = (_Float16)(d1 * rstd * g[c0+1] + b[c0+1]);
    orow[c0+2] = (_Float16)(d2 * rstd * g[c0+2] + b[c0+2]);
    orow[c0+3] = (_Float16)(d3 * rstd * g[c0+3] + b[c0+3]);
}

// ---------------------------------------------------------------------------
extern "C" void kernel_launch(void* const* d_in, const int* in_sizes, int n_in,
                              void* d_out, int out_size, void* d_ws, size_t ws_size,
                              hipStream_t stream)
{
    (void)in_sizes; (void)n_in; (void)out_size; (void)ws_size;

    const float* features = (const float*)d_in[0];
    const int*   q_ind    = (const int*)d_in[1];
    const int*   k_ind    = (const int*)d_in[2];
    const unsigned char* q_mask = (const unsigned char*)d_in[3];
    const unsigned char* k_mask = (const unsigned char*)d_in[4];
    const float* q_pos = (const float*)d_in[5];
    const float* k_pos = (const float*)d_in[6];
    const float* ln1_g = (const float*)d_in[7];
    const float* ln1_b = (const float*)d_in[8];
    const float* pw1   = (const float*)d_in[9];
    const float* pb1   = (const float*)d_in[10];
    const float* pw2   = (const float*)d_in[11];
    const float* pb2   = (const float*)d_in[12];
    const float* wq    = (const float*)d_in[13];
    const float* bq    = (const float*)d_in[14];
    const float* wk    = (const float*)d_in[15];
    const float* bk    = (const float*)d_in[16];
    const float* wv    = (const float*)d_in[17];
    const float* bv    = (const float*)d_in[18];
    const float* wo    = (const float*)d_in[19];
    const float* bo    = (const float*)d_in[20];
    const float* ln2_g = (const float*)d_in[21];
    const float* ln2_b = (const float*)d_in[22];
    const float* fw1   = (const float*)d_in[23];
    const float* fb1   = (const float*)d_in[24];
    const float* fw2   = (const float*)d_in[25];
    const float* fb2   = (const float*)d_in[26];

    // workspace carve-up (deterministic, same every call)
    char* ws = (char*)d_ws;
    size_t off = 0;
    auto alloc = [&](size_t bytes) -> void* {
        void* p = ws + off;
        off += (bytes + 255) & ~(size_t)255;
        return p;
    };
    float*    h32   = (float*)   alloc((size_t)Npt * Cc * 4);
    _Float16* qf16  = (_Float16*)alloc((size_t)PQ  * Cc * 2);
    _Float16* kt16  = (_Float16*)alloc((size_t)PK  * Cc * 2);
    _Float16* q16   = (_Float16*)alloc((size_t)PQ  * Cc * 2);
    _Float16* k16   = (_Float16*)alloc((size_t)PK  * Cc * 2);
    _Float16* v16   = (_Float16*)alloc((size_t)PK  * Cc * 2);
    _Float16* o16   = (_Float16*)alloc((size_t)PQ  * Cc * 2);
    float*    sums  = (float*)   alloc((size_t)Npt * Cc * 4);
    float*    cnt   = (float*)   alloc((size_t)Npt * 4);
    float*    y32   = (float*)   alloc((size_t)Npt * Cc * 4);
    _Float16* h2_16 = (_Float16*)alloc((size_t)Npt * Cc * 2);
    _Float16* t16   = (_Float16*)alloc((size_t)Npt * FFd * 2);
    _Float16* pw2p  = (_Float16*)alloc((size_t)Cc * Cc * 2);
    _Float16* wqp   = (_Float16*)alloc((size_t)Cc * Cc * 2);
    _Float16* wkp   = (_Float16*)alloc((size_t)Cc * Cc * 2);
    _Float16* wvp   = (_Float16*)alloc((size_t)Cc * Cc * 2);
    _Float16* wop   = (_Float16*)alloc((size_t)Cc * Cc * 2);
    _Float16* fw1p  = (_Float16*)alloc((size_t)Cc * FFd * 2);
    _Float16* fw2p  = (_Float16*)alloc((size_t)FFd * Cc * 2);

    const dim3 blk(256);

    // 0) pack weights into WMMA B-fragment layout (f16)
    pack_b_kernel<<<(Cc*Cc)/256,  blk, 0, stream>>>(pw2, pw2p, Cc, Cc);
    pack_b_kernel<<<(Cc*Cc)/256,  blk, 0, stream>>>(wq,  wqp,  Cc, Cc);
    pack_b_kernel<<<(Cc*Cc)/256,  blk, 0, stream>>>(wk,  wkp,  Cc, Cc);
    pack_b_kernel<<<(Cc*Cc)/256,  blk, 0, stream>>>(wv,  wvp,  Cc, Cc);
    pack_b_kernel<<<(Cc*Cc)/256,  blk, 0, stream>>>(wo,  wop,  Cc, Cc);
    pack_b_kernel<<<(Cc*FFd)/256, blk, 0, stream>>>(fw1, fw1p, Cc, FFd);
    pack_b_kernel<<<(FFd*Cc)/256, blk, 0, stream>>>(fw2, fw2p, FFd, Cc);

    // 1) LN1
    ln1_kernel<<<Npt/8, blk, 0, stream>>>(features, ln1_g, ln1_b, h32, Npt);

    // 2) pos-MLP + gather-add -> qf16 / kt16
    posgather_kernel<<<PQ/16, blk, 0, stream>>>(q_pos, pw1, pb1, pw2p, pb2, h32, q_ind, qf16);
    posgather_kernel<<<PK/16, blk, 0, stream>>>(k_pos, pw1, pb1, pw2p, pb2, h32, k_ind, kt16);

    // 3) Q/K/V projections (WMMA GEMM, async-LDS A staging, f16 out)
    gemm16<0,128><<<dim3(PQ/16,1), blk, 0, stream>>>(qf16, wqp, bq, q16, Cc, nullptr, nullptr, nullptr, nullptr);
    gemm16<0,128><<<dim3(PK/16,1), blk, 0, stream>>>(kt16, wkp, bk, k16, Cc, nullptr, nullptr, nullptr, nullptr);
    gemm16<0,128><<<dim3(PK/16,1), blk, 0, stream>>>(kt16, wvp, bv, v16, Cc, nullptr, nullptr, nullptr, nullptr);

    // 4) attention (wave per window-head)
    attn_kernel<<<(Wn*Hh)/8, blk, 0, stream>>>(q16, k16, v16, k_mask, o16);

    // 5) output projection + masked scatter-add into sums / cnt
    hipMemsetAsync(sums, 0, (size_t)Npt * Cc * 4, stream);
    hipMemsetAsync(cnt,  0, (size_t)Npt * 4, stream);
    gemm16<2,128><<<dim3(PQ/16,1), blk, 0, stream>>>(o16, wop, bo, nullptr, Cc, q_ind, q_mask, sums, nullptr);
    cnt_kernel<<<(PQ+255)/256, blk, 0, stream>>>(q_ind, q_mask, cnt, PQ);

    // 6) residual average + LN2
    resid_ln2_kernel<<<Npt/8, blk, 0, stream>>>(features, sums, cnt, ln2_g, ln2_b, y32, h2_16, Npt);

    // 7) FFN: (128->512 ReLU f16) then (512->128 fp32 + residual -> d_out)
    gemm16<1,128><<<dim3(Npt/16,4), blk, 0, stream>>>(h2_16, fw1p, fb1, t16, FFd, nullptr, nullptr, nullptr, nullptr);
    gemm16<3,512><<<dim3(Npt/16,1), blk, 0, stream>>>(t16, fw2p, fb2, d_out, Cc, nullptr, nullptr, nullptr, y32);
}
